// OptimizedAttention_25434796326938
// MI455X (gfx1250) — matrix operand order repaired and checked by reading, not compiled, not run on hardware
//
#include <hip/hip_runtime.h>

// MI455X (gfx1250) causal MHA: bf16 WMMA GEMMs + flash attention.
// v5: GEMM k-slice staging via Tensor Data Mover (tensor_load_to_lds,
//     TENSORcnt double-buffer); attention K staging via async-to-LDS
//     (ASYNCcnt). Fallbacks keep compile green if builtins are absent.
// Workspace (bytes): [0,8M) xb/ctx ; [8,16M) qb ; [16,24M) kb ; [24,32M) vb ;
//                    [32,34M) wqb ; [34,36) wkb ; [36,38) wvb ; [38,40) wob

#define DEV __device__ __forceinline__

typedef __attribute__((ext_vector_type(16))) unsigned short v16us;
typedef __attribute__((ext_vector_type(8)))  unsigned short v8us;
typedef __attribute__((ext_vector_type(16))) __bf16         v16bf;
typedef __attribute__((ext_vector_type(8)))  float          v8f;
typedef __attribute__((ext_vector_type(4)))  int            v4i;
typedef __attribute__((ext_vector_type(8)))  int            v8i;
typedef __attribute__((ext_vector_type(4)))  unsigned int   v4u;

typedef __attribute__((address_space(1))) v4i  g_v4i;  // global 128b chunk
typedef __attribute__((address_space(3))) v4i  l_v4i;  // LDS 128b chunk
typedef __attribute__((address_space(3))) void lds_void;

static constexpr int Bn = 2, Sn = 2048, Dn = 1024, Hn = 16, HDn = 64;
static constexpr int Mrows = Bn * Sn;  // 4096

#if defined(__has_builtin)
#if __has_builtin(__builtin_amdgcn_global_load_async_to_lds_b128)
#define HAVE_ASYNC_LDS 1
#endif
#if __has_builtin(__builtin_amdgcn_tensor_load_to_lds)
#define HAVE_TDM 1
#endif
#endif
#ifndef HAVE_ASYNC_LDS
#define HAVE_ASYNC_LDS 0
#endif
#ifndef HAVE_TDM
#define HAVE_TDM 0
#endif

#if HAVE_ASYNC_LDS
#define ASYNC_WAIT(n) asm volatile("s_wait_asynccnt " #n ::: "memory")
#else
#define ASYNC_WAIT(n) do {} while (0)
#endif

// copy 16 bytes global -> LDS (async engine when available)
DEV void cp16_g2l(unsigned short* l, const unsigned short* g) {
#if HAVE_ASYNC_LDS
  __builtin_amdgcn_global_load_async_to_lds_b128((g_v4i*)g, (l_v4i*)l, 0, 0);
#else
  *(v8us*)l = *(const v8us*)g;
#endif
}

DEV unsigned lds_addr_of(const void* p) {  // LDS byte offset of a shared ptr
  return (unsigned)(unsigned long long)(lds_void*)p;
}

#if HAVE_TDM
// 2D tile DMA: global (row-major, row_stride elems of 2B) -> contiguous LDS.
// Tensor dims == tile dims (tiles are always fully in-bounds here).
DEV void tdm_load_2d(unsigned lds_off, const unsigned short* gaddr,
                     int tile_w, int tile_h, int row_stride) {
  const unsigned long long ga = (unsigned long long)(size_t)gaddr;
  v4u g0;
  g0[0] = 1u;                                  // count=1, user D#
  g0[1] = lds_off;                             // lds_addr [63:32]
  g0[2] = (unsigned)(ga & 0xFFFFFFFFu);        // global_addr lo
  g0[3] = (unsigned)((ga >> 32) & 0x01FFFFFFu) // global_addr [56:32]
          | (2u << 30);                        // type=2 ("image")
  v8i g1;
  g1[0] = (int)(1u << 16);                     // data_size=1 (2 bytes)
  g1[1] = (int)(((unsigned)tile_w & 0xFFFFu) << 16);   // tensor_dim0 lo16
  g1[2] = (int)(((unsigned)tile_h & 0xFFFFu) << 16);   // dim0 hi | tensor_dim1 lo16
  g1[3] = (int)(((unsigned)tile_w & 0xFFFFu) << 16);   // dim1 hi | tile_dim0
  g1[4] = (int)((unsigned)tile_h & 0xFFFFu);           // tile_dim1 | tile_dim2=0
  g1[5] = row_stride;                          // tensor_dim0_stride lo32
  g1[6] = 0;                                   // stride hi | dim1_stride lo
  g1[7] = 0;
  const v4i z4 = {0, 0, 0, 0};                 // groups 2/3: 2-D tensor
  const v8i z8 = {0, 0, 0, 0, 0, 0, 0, 0};
  __builtin_amdgcn_tensor_load_to_lds(g0, g1, z4, z4, z8, 0);
}
#endif

DEV unsigned short f2bf(float f) {
  unsigned u = __builtin_bit_cast(unsigned, f);
  u += 0x7FFFu + ((u >> 16) & 1u);  // round-to-nearest-even
  return (unsigned short)(u >> 16);
}
DEV v16bf bc16(v16us v) { return __builtin_bit_cast(v16bf, v); }

// A fragment 16x32 bf16 (contiguous k, same order as B): lane L -> m=L&15, hi=L>>4;
// elems 0..7 = K[hi*16..+7], elems 8..15 = K[hi*16+8..+7]
DEV v16bf load_a_frag(const unsigned short* base, int ld, int lane) {
  const int m = lane & 15, hi = lane >> 4;
  const unsigned short* p = base + (size_t)m * ld + hi * 16;
  v8us lo = *(const v8us*)p;
  v8us hf = *(const v8us*)(p + 8);
  v16us r;
#pragma unroll
  for (int i = 0; i < 8; ++i) { r[i] = lo[i]; r[i + 8] = hf[i]; }
  return bc16(r);
}

// B fragment 32x16 bf16: lane L -> n=L&15, hi=L>>4; K = hi*16..+15 contiguous.
// base_nk is row-major [N,K] (= mathematical B transposed = torch W layout).
DEV v16bf load_b_frag(const unsigned short* base_nk, int ld, int lane) {
  const int n = lane & 15, hi = lane >> 4;
  return bc16(*(const v16us*)(base_nk + (size_t)n * ld + hi * 16));
}

DEV v8f wmma_bf16(v16bf a, v16bf b, v8f c) {
  return __builtin_amdgcn_wmma_f32_16x16x32_bf16(false, a, false, b, (short)0,
                                                 c, false, false);
}

// ---------------------------------------------------------------- convert
__global__ void k_cvt(const float* __restrict__ src,
                      unsigned short* __restrict__ dst, int n) {
  int i = blockIdx.x * blockDim.x + threadIdx.x;
  const int stride = gridDim.x * blockDim.x;
  for (; i < n; i += stride) dst[i] = f2bf(src[i]);
}

// ---------------------------------------------------------------- GEMM
// C[m,n] = sum_k A[m,k]*W[n,k].  Block 256 thr = 8 waves (2x4).
// Block tile 128x256, wave tile 64x64 (4x4 WMMA accums).
// K sliced by 32; slices DMA'd by the TDM into double-buffered LDS.
template <bool OUTF32>
__global__ void __launch_bounds__(256)
k_gemm(const unsigned short* __restrict__ A,
       const unsigned short* __restrict__ W,
       void* __restrict__ Cout, int M, int N, int K) {
  __shared__ unsigned short sA[2][128 * 32];  // 2 x 8KB
  __shared__ unsigned short sB[2][256 * 32];  // 2 x 16KB
  const int tid = threadIdx.x;
  const int lane = tid & 31, w = tid >> 5;
  const int bm0 = blockIdx.x * 128, bn0 = blockIdx.y * 256;
  const int wm = (w >> 2) * 64, wn = (w & 3) * 64;
#if !HAVE_TDM
  const int srow = tid >> 2;       // 0..63
  const int scol = (tid & 3) * 8;  // 0,8,16,24
#endif

  v8f acc[4][4] = {};

  auto stage = [&](int k0, int buf) {
#if HAVE_TDM
    if (w == 0) {  // one wave describes both tiles; TDM does the walk
      tdm_load_2d(lds_addr_of(&sA[buf][0]), A + (size_t)bm0 * K + k0,
                  /*w=*/32, /*h=*/128, /*stride=*/K);
      tdm_load_2d(lds_addr_of(&sB[buf][0]), W + (size_t)bn0 * K + k0,
                  /*w=*/32, /*h=*/256, /*stride=*/K);
    }
#else
#pragma unroll
    for (int i = 0; i < 2; ++i) {
      const int row = srow + i * 64;
      cp16_g2l(&sA[buf][row * 32 + scol],
               A + (size_t)(bm0 + row) * K + k0 + scol);
    }
#pragma unroll
    for (int i = 0; i < 4; ++i) {
      const int row = srow + i * 64;
      cp16_g2l(&sB[buf][row * 32 + scol],
               W + (size_t)(bn0 + row) * K + k0 + scol);
    }
#endif
  };

  stage(0, 0);
  const int nk = K / 32;
  for (int kt = 0; kt < nk; ++kt) {
    const int buf = kt & 1;
    if (kt + 1 < nk) {
      stage((kt + 1) * 32, buf ^ 1);  // prefetch next slice into alt buffer
#if HAVE_TDM
      if (w == 0) __builtin_amdgcn_s_wait_tensorcnt(2);  // drain current 2
#else
      ASYNC_WAIT(0x6);  // drain the 6 ops of the current slice
#endif
    } else {
#if HAVE_TDM
      if (w == 0) __builtin_amdgcn_s_wait_tensorcnt(0);
#else
      ASYNC_WAIT(0x0);
#endif
    }
    __syncthreads();
    v16bf af[4], bfr[4];
#pragma unroll
    for (int i = 0; i < 4; ++i)
      af[i] = load_a_frag(&sA[buf][(wm + i * 16) * 32], 32, lane);
#pragma unroll
    for (int j = 0; j < 4; ++j)
      bfr[j] = load_b_frag(&sB[buf][(wn + j * 16) * 32], 32, lane);
#pragma unroll
    for (int i = 0; i < 4; ++i)
#pragma unroll
      for (int j = 0; j < 4; ++j)
        acc[i][j] = wmma_bf16(af[i], bfr[j], acc[i][j]);
    __syncthreads();
  }

  const int hi = lane >> 4, n = lane & 15;
#pragma unroll
  for (int i = 0; i < 4; ++i)
#pragma unroll
    for (int j = 0; j < 4; ++j)
#pragma unroll
      for (int r = 0; r < 8; ++r) {
        const size_t mi = bm0 + wm + i * 16 + r + 8 * hi;
        const size_t ni = bn0 + wn + j * 16 + n;
        const float v = acc[i][j][r];
        if constexpr (OUTF32) ((float*)Cout)[mi * N + ni] = v;
        else ((unsigned short*)Cout)[mi * N + ni] = f2bf(v);
      }
}

// ---------------------------------------------------------------- attention
// grid (S/128, B*H); block 256 = 8 waves; each wave owns 16 q-rows.
__global__ void __launch_bounds__(256)
k_attn(const unsigned short* __restrict__ Qg,
       const unsigned short* __restrict__ Kg,
       const unsigned short* __restrict__ Vg,
       unsigned short* __restrict__ Ctx) {
  __shared__ unsigned short sK[32 * 64];     // [key][d]
  __shared__ unsigned short sV[64 * 32];     // [d][key] (transposed)
  __shared__ unsigned short sP[8][16 * 32];  // per-wave P staging

  const int tid = threadIdx.x;
  const int lane = tid & 31, w = tid >> 5;
  const int b = blockIdx.y >> 4, h = blockIdx.y & 15;
  const int q0 = blockIdx.x * 128;
  const int qm = q0 + w * 16;
  const int hi = lane >> 4, nn = lane & 15;
  const size_t qbase = ((size_t)b * Sn + qm) * Dn + h * HDn;

  const v16bf aq0 = load_a_frag(Qg + qbase, Dn, lane);       // d 0..31
  const v16bf aq1 = load_a_frag(Qg + qbase + 32, Dn, lane);  // d 32..63

  v8f accC[4] = {};
  float rmax[8], rsum[8];
#pragma unroll
  for (int r = 0; r < 8; ++r) { rmax[r] = -1e30f; rsum[r] = 0.f; }

  const int srow = tid >> 3;       // key row 0..31
  const int scol = (tid & 7) * 8;  // d 0,8..56
  const int kv_end = q0 + 128;     // causal bound for this q-block

  for (int kidx = 0; kidx < kv_end; kidx += 32) {
    const size_t gk = ((size_t)b * Sn + kidx + srow) * Dn + h * HDn + scol;
    cp16_g2l(&sK[srow * 64 + scol], Kg + gk);  // async K tile
    v8us vv = *(const v8us*)(Vg + gk);         // V tile, transposed by hand
#pragma unroll
    for (int j = 0; j < 8; ++j) sV[(scol + j) * 32 + srow] = vv[j];
    ASYNC_WAIT(0x0);
    __syncthreads();

    // scores S = Q K^T (two 16-key subtiles, d=64 in 2 chunks)
    v8f sc[2];
#pragma unroll
    for (int kc = 0; kc < 2; ++kc) {
      v16bf bk0 = load_b_frag(&sK[kc * 16 * 64], 64, lane);
      v16bf bk1 = load_b_frag(&sK[kc * 16 * 64 + 32], 64, lane);
      v8f s = {};
      s = wmma_bf16(aq0, bk0, s);
      s = wmma_bf16(aq1, bk1, s);
      sc[kc] = s;
    }

    // online softmax, common max across both subtiles of this kv step
#pragma unroll
    for (int r = 0; r < 8; ++r) {
      const int qg = qm + r + 8 * hi;
      float s0 = sc[0][r] * 0.125f;  // 1/sqrt(64)
      float s1 = sc[1][r] * 0.125f;
      if (kidx + nn > qg) s0 = -1e30f;  // causal mask
      if (kidx + 16 + nn > qg) s1 = -1e30f;
      float tmax = fmaxf(s0, s1);
      tmax = fmaxf(tmax, __shfl_xor(tmax, 1, 32));
      tmax = fmaxf(tmax, __shfl_xor(tmax, 2, 32));
      tmax = fmaxf(tmax, __shfl_xor(tmax, 4, 32));
      tmax = fmaxf(tmax, __shfl_xor(tmax, 8, 32));
      const float mnew = fmaxf(rmax[r], tmax);
      const float alpha = __expf(rmax[r] - mnew);
      const float p0 = __expf(s0 - mnew);
      const float p1 = __expf(s1 - mnew);
      float ps = p0 + p1;
      ps += __shfl_xor(ps, 1, 32);
      ps += __shfl_xor(ps, 2, 32);
      ps += __shfl_xor(ps, 4, 32);
      ps += __shfl_xor(ps, 8, 32);
      rsum[r] = rsum[r] * alpha + ps;
      rmax[r] = mnew;
#pragma unroll
      for (int t = 0; t < 4; ++t) accC[t][r] *= alpha;
      sP[w][(r + 8 * hi) * 32 + nn] = f2bf(p0);
      sP[w][(r + 8 * hi) * 32 + 16 + nn] = f2bf(p1);
    }
    asm volatile("s_wait_dscnt 0" ::: "memory");  // same-wave LDS RAW

    // ctx += P @ V
    const v16bf ap = load_a_frag(&sP[w][0], 32, lane);
#pragma unroll
    for (int t = 0; t < 4; ++t) {
      v16bf bv = load_b_frag(&sV[t * 16 * 32], 32, lane);
      accC[t] = wmma_bf16(ap, bv, accC[t]);
    }
    __syncthreads();
  }

#pragma unroll
  for (int t = 0; t < 4; ++t)
#pragma unroll
    for (int r = 0; r < 8; ++r) {
      const float v = accC[t][r] / rsum[r];
      Ctx[((size_t)b * Sn + qm + r + 8 * hi) * Dn + h * HDn + t * 16 + nn] =
          f2bf(v);
    }
}

// ---------------------------------------------------------------- launch
extern "C" void kernel_launch(void* const* d_in, const int* in_sizes, int n_in,
                              void* d_out, int out_size, void* d_ws,
                              size_t ws_size, hipStream_t stream) {
  (void)in_sizes; (void)n_in; (void)out_size; (void)ws_size;
  const float* x  = (const float*)d_in[0];
  const float* wq = (const float*)d_in[1];
  const float* wk = (const float*)d_in[2];
  const float* wv = (const float*)d_in[3];
  const float* wo = (const float*)d_in[4];

  char* ws = (char*)d_ws;
  const size_t MB = 1024 * 1024;
  unsigned short* xb  = (unsigned short*)(ws + 0);  // also ctx later
  unsigned short* qb  = (unsigned short*)(ws + 8 * MB);
  unsigned short* kb  = (unsigned short*)(ws + 16 * MB);
  unsigned short* vb  = (unsigned short*)(ws + 24 * MB);
  unsigned short* wqb = (unsigned short*)(ws + 32 * MB);
  unsigned short* wkb = (unsigned short*)(ws + 34 * MB);
  unsigned short* wvb = (unsigned short*)(ws + 36 * MB);
  unsigned short* wob = (unsigned short*)(ws + 38 * MB);

  const int nx = Mrows * Dn;  // 4,194,304
  const int nw = Dn * Dn;     // 1,048,576
  k_cvt<<<1024, 256, 0, stream>>>(x, xb, nx);
  k_cvt<<<512, 256, 0, stream>>>(wq, wqb, nw);
  k_cvt<<<512, 256, 0, stream>>>(wk, wkb, nw);
  k_cvt<<<512, 256, 0, stream>>>(wv, wvb, nw);
  k_cvt<<<512, 256, 0, stream>>>(wo, wob, nw);

  dim3 gg(Mrows / 128, Dn / 256);  // (32, 4)
  k_gemm<false><<<gg, 256, 0, stream>>>(xb, wqb, qb, Mrows, Dn, Dn);
  k_gemm<false><<<gg, 256, 0, stream>>>(xb, wkb, kb, Mrows, Dn, Dn);
  k_gemm<false><<<gg, 256, 0, stream>>>(xb, wvb, vb, Mrows, Dn, Dn);

  dim3 ga(Sn / 128, Bn * Hn);  // (16, 32)
  k_attn<<<ga, 256, 0, stream>>>(qb, kb, vb, xb);

  k_gemm<true><<<gg, 256, 0, stream>>>(xb, wob, d_out, Mrows, Dn, Dn);
}
